// QuantizeAttention_5875515261025
// MI455X (gfx1250) — matrix operand order repaired and checked by reading, not compiled, not run on hardware
//
#include <hip/hip_runtime.h>
#include <hip/hip_bf16.h>

typedef __bf16 bf16;
typedef __attribute__((ext_vector_type(16))) __bf16 v16bf;
typedef __attribute__((ext_vector_type(8)))  float  v8f;

#define NHEAD 12
#define NTOK  197
#define DIM   64
#define CDIM  768
#define MROWS 12608          // 64*197
#define NCODE 50

__device__ __forceinline__ v8f wmma_bf16(v16bf a, v16bf b, v8f c) {
  return __builtin_amdgcn_wmma_f32_16x16x32_bf16(false, a, false, b, (short)0, c, false, false);
}

// ---- fragment loaders (contiguous k = 16*half + s, same in A and B) -----------------------
// A 16x32 bf16: lane%16 = row; lanes0-15 hold K = 0..15, lanes16-31 K = 16..31
__device__ __forceinline__ v16bf load_a16x32(const bf16* p, int ld) {
  int lane = threadIdx.x & 31; int r = lane & 15, half = lane >> 4;
  const bf16* q = p + r * ld + half * 16;
  v16bf a; uint4* d = (uint4*)&a;
  d[0] = *(const uint4*)(q);
  d[1] = *(const uint4*)(q + 8);
  return a;
}
__device__ __forceinline__ v16bf load_a16x32_guard(const bf16* p, int ld, int nrows) {
  int lane = threadIdx.x & 31; int r = lane & 15, half = lane >> 4;
  v16bf a; uint4* d = (uint4*)&a;
  if (r < nrows) {
    const bf16* q = p + r * ld + half * 16;
    d[0] = *(const uint4*)(q);
    d[1] = *(const uint4*)(q + 8);
  } else {
    d[0] = uint4{0,0,0,0}; d[1] = uint4{0,0,0,0};
  }
  return a;
}
// B 32x16 where B[k][n] = p[n*ld + k] (i.e. "columns" stored contiguously in k)
__device__ __forceinline__ v16bf load_bNT(const bf16* p, int ld) {
  int lane = threadIdx.x & 31; int n = lane & 15, half = lane >> 4;
  const bf16* q = p + n * ld + half * 16;
  v16bf b; uint4* d = (uint4*)&b;
  d[0] = *(const uint4*)(q);
  d[1] = *(const uint4*)(q + 8);
  return b;
}
__device__ __forceinline__ v16bf load_bNT_guard(const bf16* p, int ld, int ncols) {
  int lane = threadIdx.x & 31; int n = lane & 15, half = lane >> 4;
  v16bf b; uint4* d = (uint4*)&b;
  if (n < ncols) {
    const bf16* q = p + n * ld + half * 16;
    d[0] = *(const uint4*)(q);
    d[1] = *(const uint4*)(q + 8);
  } else {
    d[0] = uint4{0,0,0,0}; d[1] = uint4{0,0,0,0};
  }
  return b;
}
// B 32x16 where B[k][n] = p[k*ld + n] (row-major K x N), guard k < nk
__device__ __forceinline__ v16bf load_bT_guard(const bf16* p, int ld, int k0, int nk) {
  int lane = threadIdx.x & 31; int n = lane & 15, half = lane >> 4;
  v16bf b; int kk = k0 + half * 16;
  #pragma unroll
  for (int i = 0; i < 16; ++i) {
    int k = kk + i;
    b[i] = (k < nk) ? p[k * ld + n] : (bf16)0.f;
  }
  return b;
}

// ---- kernels ---------------------------------------------------------------
__global__ void cvt_bf16(const float* __restrict__ s, bf16* __restrict__ d, int n) {
  int i = blockIdx.x * blockDim.x + threadIdx.x;
  if (i < n) d[i] = (bf16)s[i];
}

// out[m,o] = sum_c x[m,c]*w[o,c]; scatter to q/k/v in [B,H,N,d] bf16
__global__ __launch_bounds__(256) void qkv_gemm(
    const bf16* __restrict__ xb, const bf16* __restrict__ wb,
    bf16* __restrict__ qb, bf16* __restrict__ kb, bf16* __restrict__ vb) {
  const int lane = threadIdx.x & 31, wave = threadIdx.x >> 5;
  const int m0 = (blockIdx.x * 2 + (wave >> 2)) * 16;
  const int n0 = blockIdx.y * 256 + (wave & 3) * 64;
  v8f acc[4];
  #pragma unroll
  for (int t = 0; t < 4; ++t) acc[t] = (v8f)0.0f;
  for (int kt = 0; kt < CDIM / 32; ++kt) {
    v16bf a = load_a16x32(xb + m0 * CDIM + kt * 32, CDIM);
    #pragma unroll
    for (int t = 0; t < 4; ++t) {
      v16bf b = load_bNT(wb + (n0 + t * 16) * CDIM + kt * 32, CDIM);
      acc[t] = wmma_bf16(a, b, acc[t]);
    }
  }
  const int nn = lane & 15, half = lane >> 4;
  #pragma unroll
  for (int t = 0; t < 4; ++t) {
    int o = n0 + t * 16 + nn;
    int s = o / CDIM; int rem = o - s * CDIM;
    int h = rem >> 6; int dd = rem & 63;
    bf16* dst = (s == 0) ? qb : ((s == 1) ? kb : vb);
    #pragma unroll
    for (int r = 0; r < 8; ++r) {
      int row = m0 + r + 8 * half;
      int bb = row / NTOK; int tok = row - bb * NTOK;
      dst[((bb * NHEAD + h) * NTOK + tok) * DIM + dd] = (bf16)acc[t][r];
    }
  }
}

// one wave per token: nearest-codebook quantization + MSE partials
__global__ __launch_bounds__(256) void quantize_kernel(
    const bf16* __restrict__ qb, const bf16* __restrict__ kb,
    const float* __restrict__ qe, const float* __restrict__ ke,
    bf16* __restrict__ catq, bf16* __restrict__ catk,
    float* __restrict__ msepart) {
  const int lane = threadIdx.x & 31, wave = threadIdx.x >> 5;
  const int which = blockIdx.y;
  const int t = blockIdx.x * 8 + wave;               // 0 .. 151295
  const bf16* src = (which ? kb : qb) + t * DIM;
  bf16*       dst = (which ? catk : catq) + t * DIM;
  const int tok = t % NTOK;
  const int h = (t / NTOK) % NHEAD;
  float part = 0.f;
  if (tok == 0) {                                    // cls row passes through
    dst[lane * 2]     = src[lane * 2];
    dst[lane * 2 + 1] = src[lane * 2 + 1];
  } else {
    const float* emb = (which ? ke : qe) + h * NCODE * DIM;
    float f0 = (float)src[lane * 2], f1 = (float)src[lane * 2 + 1];
    float bestd = 3.4e38f; int besti = 0;
    for (int c = 0; c < NCODE; ++c) {
      float e0 = emb[c * DIM + lane * 2], e1 = emb[c * DIM + lane * 2 + 1];
      float d0 = f0 - e0, d1 = f1 - e1;
      float ds = d0 * d0 + d1 * d1;
      #pragma unroll
      for (int off = 16; off; off >>= 1) ds += __shfl_xor(ds, off);
      if (ds < bestd) { bestd = ds; besti = c; }     // first-min == argmin
    }
    float q0 = emb[besti * DIM + lane * 2], q1 = emb[besti * DIM + lane * 2 + 1];
    dst[lane * 2]     = (bf16)q0;
    dst[lane * 2 + 1] = (bf16)q1;
    float d0 = q0 - f0, d1 = q1 - f1;
    part = d0 * d0 + d1 * d1;
    #pragma unroll
    for (int off = 16; off; off >>= 1) part += __shfl_xor(part, off);
  }
  __shared__ float wsum[8];
  if (lane == 0) wsum[wave] = part;
  __syncthreads();
  if (threadIdx.x == 0) {
    float s = 0.f;
    for (int i = 0; i < 8; ++i) s += wsum[i];        // fixed order -> deterministic
    msepart[blockIdx.y * 18912 + blockIdx.x] = s;
  }
}

// one wave per (b,h,16-row tile): quant softmax -> LDS, orig softmax + KL, P@V
__global__ __launch_bounds__(32) void attn_kernel(
    const bf16* __restrict__ qb, const bf16* __restrict__ kb, const bf16* __restrict__ vb,
    const bf16* __restrict__ catq, const bf16* __restrict__ catk,
    bf16* __restrict__ obuf, float* __restrict__ klpart) {
  const int rt = blockIdx.x, h = blockIdx.y, b = blockIdx.z;
  const int bh = b * NHEAD + h;
  const int base = bh * NTOK * DIM;
  const bf16 *qh = qb + base, *kh = kb + base, *vh = vb + base;
  const bf16 *cqh = catq + base, *ckh = catk + base;
  const int lane = threadIdx.x;
  const int nn = lane & 15, half = lane >> 4;
  const int row0 = rt * 16;
  const float scale = 0.125f;                         // d^-0.5, d=64
  __shared__ bf16 pq[16 * 232];                       // P_quant, padded to 224 cols

  v8f acc[13];
  // ---- pass A: quantized scores + softmax -> LDS ----
  #pragma unroll
  for (int ct = 0; ct < 13; ++ct) acc[ct] = (v8f)0.0f;
  for (int kt = 0; kt < 2; ++kt) {
    v16bf a = load_a16x32_guard(cqh + row0 * DIM + kt * 32, DIM, NTOK - row0);
    #pragma unroll
    for (int ct = 0; ct < 13; ++ct) {
      v16bf bb = load_bNT_guard(ckh + ct * 16 * DIM + kt * 32, DIM, NTOK - ct * 16);
      acc[ct] = wmma_bf16(a, bb, acc[ct]);
    }
  }
  #pragma unroll
  for (int r = 0; r < 8; ++r) {
    float mx = -1e30f;
    #pragma unroll
    for (int ct = 0; ct < 13; ++ct) {
      float v = acc[ct][r] * scale;
      if (ct * 16 + nn >= NTOK) v = -1e30f;
      acc[ct][r] = v; mx = fmaxf(mx, v);
    }
    #pragma unroll
    for (int off = 1; off < 16; off <<= 1) mx = fmaxf(mx, __shfl_xor(mx, off));
    float sum = 0.f;
    #pragma unroll
    for (int ct = 0; ct < 13; ++ct) {
      float e = (ct * 16 + nn < NTOK) ? __expf(acc[ct][r] - mx) : 0.f;
      acc[ct][r] = e; sum += e;
    }
    #pragma unroll
    for (int off = 1; off < 16; off <<= 1) sum += __shfl_xor(sum, off);
    float inv = 1.f / sum;
    int rr = r + 8 * half;
    #pragma unroll
    for (int ct = 0; ct < 13; ++ct)
      pq[rr * 232 + ct * 16 + nn] = (bf16)(acc[ct][r] * inv);
  }
  #pragma unroll
  for (int r = 0; r < 8; ++r)                         // zero pad cols 208..223
    pq[(r * 2 + half) * 232 + 208 + nn] = (bf16)0.f;
  __syncthreads();

  // ---- pass B: original scores + softmax, KL(p_orig || uses p_quant) ----
  #pragma unroll
  for (int ct = 0; ct < 13; ++ct) acc[ct] = (v8f)0.0f;
  for (int kt = 0; kt < 2; ++kt) {
    v16bf a = load_a16x32_guard(qh + row0 * DIM + kt * 32, DIM, NTOK - row0);
    #pragma unroll
    for (int ct = 0; ct < 13; ++ct) {
      v16bf bb = load_bNT_guard(kh + ct * 16 * DIM + kt * 32, DIM, NTOK - ct * 16);
      acc[ct] = wmma_bf16(a, bb, acc[ct]);
    }
  }
  float kl = 0.f;
  #pragma unroll
  for (int r = 0; r < 8; ++r) {
    float mx = -1e30f;
    #pragma unroll
    for (int ct = 0; ct < 13; ++ct) {
      float v = acc[ct][r] * scale;
      if (ct * 16 + nn >= NTOK) v = -1e30f;
      acc[ct][r] = v; mx = fmaxf(mx, v);
    }
    #pragma unroll
    for (int off = 1; off < 16; off <<= 1) mx = fmaxf(mx, __shfl_xor(mx, off));
    float sum = 0.f;
    #pragma unroll
    for (int ct = 0; ct < 13; ++ct) {
      float e = (ct * 16 + nn < NTOK) ? __expf(acc[ct][r] - mx) : 0.f;
      acc[ct][r] = e; sum += e;
    }
    #pragma unroll
    for (int off = 1; off < 16; off <<= 1) sum += __shfl_xor(sum, off);
    float inv = 1.f / sum;
    int rr = r + 8 * half;
    bool rowok = (row0 + rr) < NTOK;
    #pragma unroll
    for (int ct = 0; ct < 13; ++ct) {
      int col = ct * 16 + nn;
      if (rowok && col < NTOK) {
        float p = acc[ct][r] * inv;
        float pqv = (float)pq[rr * 232 + col];
        kl += p * (__logf(p) - pqv);
      }
    }
  }
  #pragma unroll
  for (int off = 1; off < 32; off <<= 1) kl += __shfl_xor(kl, off);
  if (lane == 0) klpart[bh * 13 + rt] = kl;

  // ---- pass C: O = P_quant @ V ----
  v8f oacc[4];
  #pragma unroll
  for (int nt = 0; nt < 4; ++nt) oacc[nt] = (v8f)0.0f;
  for (int kt = 0; kt < 7; ++kt) {
    v16bf a = load_a16x32(pq + kt * 32, 232);         // A fragment from LDS
    #pragma unroll
    for (int nt = 0; nt < 4; ++nt) {
      v16bf bb = load_bT_guard(vh + nt * 16, DIM, kt * 32, NTOK);
      oacc[nt] = wmma_bf16(a, bb, oacc[nt]);
    }
  }
  #pragma unroll
  for (int nt = 0; nt < 4; ++nt)
    #pragma unroll
    for (int r = 0; r < 8; ++r) {
      int row = row0 + r + 8 * half;
      if (row < NTOK)
        obuf[(b * NTOK + row) * CDIM + h * DIM + nt * 16 + nn] = (bf16)oacc[nt][r];
    }
}

// out[m,o] = sum_c O[m,c]*Wp[o,c] + bias[o]
__global__ __launch_bounds__(256) void proj_gemm(
    const bf16* __restrict__ ob, const bf16* __restrict__ wb,
    const float* __restrict__ bias, float* __restrict__ out) {
  const int lane = threadIdx.x & 31, wave = threadIdx.x >> 5;
  const int m0 = (blockIdx.x * 2 + (wave >> 2)) * 16;
  const int n0 = blockIdx.y * 256 + (wave & 3) * 64;
  v8f acc[4];
  #pragma unroll
  for (int t = 0; t < 4; ++t) acc[t] = (v8f)0.0f;
  for (int kt = 0; kt < CDIM / 32; ++kt) {
    v16bf a = load_a16x32(ob + m0 * CDIM + kt * 32, CDIM);
    #pragma unroll
    for (int t = 0; t < 4; ++t) {
      v16bf b = load_bNT(wb + (n0 + t * 16) * CDIM + kt * 32, CDIM);
      acc[t] = wmma_bf16(a, b, acc[t]);
    }
  }
  const int nn = lane & 15, half = lane >> 4;
  #pragma unroll
  for (int t = 0; t < 4; ++t) {
    int o = n0 + t * 16 + nn;
    float bv = bias[o];
    #pragma unroll
    for (int r = 0; r < 8; ++r) {
      int row = m0 + r + 8 * half;
      out[row * CDIM + o] = acc[t][r] + bv;
    }
  }
}

// deterministic final reduction of MSE and KL partials
__global__ void loss_reduce(const float* __restrict__ msepart,
                            const float* __restrict__ klpart,
                            float* __restrict__ loss_out) {
  __shared__ float shm[256];
  __shared__ float shk[256];
  int tid = threadIdx.x;
  float s = 0.f;
  for (int i = tid; i < 37824; i += 256) s += msepart[i];
  float k = 0.f;
  for (int i = tid; i < 9984; i += 256) k += klpart[i];
  shm[tid] = s; shk[tid] = k;
  __syncthreads();
  if (tid == 0) {
    float ms = 0.f, ks = 0.f;
    for (int i = 0; i < 256; ++i) { ms += shm[i]; ks += shk[i]; }
    loss_out[0] = ms / 9633792.0f + ks / 29805312.0f;
  }
}

extern "C" void kernel_launch(void* const* d_in, const int* in_sizes, int n_in,
                              void* d_out, int out_size, void* d_ws, size_t ws_size,
                              hipStream_t stream) {
  const float* x      = (const float*)d_in[0];
  const float* qkv_w  = (const float*)d_in[1];
  const float* proj_w = (const float*)d_in[2];
  const float* proj_b = (const float*)d_in[3];
  const float* q_emb  = (const float*)d_in[4];
  const float* k_emb  = (const float*)d_in[5];
  float* out = (float*)d_out;

  char* ws = (char*)d_ws;
  size_t off = 0;
  auto alloc = [&](size_t bytes) -> char* {
    char* p = ws + off;
    off += (bytes + 255) & ~(size_t)255;
    return p;
  };
  const size_t BHND = (size_t)MROWS * CDIM * sizeof(bf16);     // 64*12*197*64 elems too
  bf16* xb    = (bf16*)alloc(BHND);
  bf16* wqkv  = (bf16*)alloc((size_t)3 * CDIM * CDIM * sizeof(bf16));
  bf16* wproj = (bf16*)alloc((size_t)CDIM * CDIM * sizeof(bf16));
  bf16* qb    = (bf16*)alloc(BHND);
  bf16* kb    = (bf16*)alloc(BHND);
  bf16* vb    = (bf16*)alloc(BHND);
  bf16* catq  = (bf16*)alloc(BHND);
  bf16* catk  = (bf16*)alloc(BHND);
  bf16* obuf  = (bf16*)alloc(BHND);
  float* msepart = (float*)alloc(37824 * sizeof(float));
  float* klpart  = (float*)alloc(9984 * sizeof(float));

  const int nx = MROWS * CDIM, nw = 3 * CDIM * CDIM, np = CDIM * CDIM;
  cvt_bf16<<<(nx + 255) / 256, 256, 0, stream>>>(x, xb, nx);
  cvt_bf16<<<(nw + 255) / 256, 256, 0, stream>>>(qkv_w, wqkv, nw);
  cvt_bf16<<<(np + 255) / 256, 256, 0, stream>>>(proj_w, wproj, np);

  qkv_gemm<<<dim3(394, 9), 256, 0, stream>>>(xb, wqkv, qb, kb, vb);
  quantize_kernel<<<dim3(18912, 2), 256, 0, stream>>>(qb, kb, q_emb, k_emb,
                                                      catq, catk, msepart);
  attn_kernel<<<dim3(13, NHEAD, 64), 32, 0, stream>>>(qb, kb, vb, catq, catk,
                                                      obuf, klpart);
  proj_gemm<<<dim3(394, 3), 256, 0, stream>>>(obuf, wproj, proj_b, out);
  loss_reduce<<<1, 256, 0, stream>>>(msepart, klpart, out + (size_t)MROWS * CDIM);
}
